// Transformer_29343216566753
// MI455X (gfx1250) — compile-verified, matrix operand order checked by reading, not run on hardware
//
#include <hip/hip_runtime.h>

// ---------------------------------------------------------------- constants
#define Bb 4
#define Ss 1000
#define Dd 512
#define Hh 8
#define Ll 4
#define Vv 32000
#define HD 64
#define FFd 2048
#define ROWS (Bb * Ss)   // 4000
#define MPAD 4032        // ROWS padded to 64 for transposed-V stores

typedef __attribute__((ext_vector_type(16))) __bf16 bf16x16;
typedef __attribute__((ext_vector_type(8)))  float  f32x8;

#define DEV static __device__ __forceinline__

// k-index inside a 32-wide K slab for 16-bit A/B WMMA fragments (ISA 7.12.2):
// element e lives in VGPR v=e>>1; K = (v/4)*16 + half*8 + (v%4)*2 + (e&1)
DEV int frag_k(int e, int half) {
    int v = e >> 1;
    return ((v >> 2) << 4) + (half << 3) + ((v & 3) << 1) + (e & 1);
}

// A fragment (16xK tile, row-major LDS tile, leading dim ldk).
// Per lane this is two contiguous 16B chunks -> ds_load_b128 x2.
DEV bf16x16 load_fragA(const __bf16* tile, int ldk, int mBase, int lane) {
    int m = mBase + (lane & 15);
    int half = lane >> 4;
    const __bf16* row = tile + m * ldk;
    bf16x16 f;
#pragma unroll
    for (int e = 0; e < 16; ++e) f[e] = row[frag_k(e, half)];
    return f;
}

// ---------------------------------------------------------------- weight pack
// fp32 row-major K x N  ->  bf16 WMMA-B-fragment-major blocks:
// block(nt, ks) = 512 bf16 (1 KB), lane-major: lane*16 + e, so a wave's
// B fragment is one contiguous 32B chunk per lane.
__global__ void k_pack_b(const float* __restrict__ src, __bf16* __restrict__ dst,
                         int K, int N) {
    int KS = K >> 5;
    long long total = (long long)K * N;
    long long i = (long long)blockIdx.x * blockDim.x + threadIdx.x;
    long long stride = (long long)gridDim.x * blockDim.x;
    for (; i < total; i += stride) {
        int within = (int)(i & 511);
        long long blk = i >> 9;
        int lane = within >> 4, e = within & 15;
        int nt = (int)(blk / KS), ks = (int)(blk % KS);
        int k = ks * 32 + frag_k(e, lane >> 4);
        int n = nt * 16 + (lane & 15);
        dst[i] = (__bf16)src[(size_t)k * N + n];
    }
}

// ---------------------------------------------------------------- embedding + PE
__global__ __launch_bounds__(128)
void k_embed(const int* __restrict__ tok, const float* __restrict__ emb,
             float* __restrict__ x, __bf16* __restrict__ xb) {
    int row = blockIdx.x;            // 0..ROWS-1
    int s = row % Ss;
    int t = tok[row];
    const float c = -9.210340371976184f / (float)Dd;   // -ln(10000)/D
    for (int d = threadIdx.x; d < Dd; d += blockDim.x) {
        float f = __expf((float)(d & ~1) * c);
        float ang = (float)s * f;
        float pe = (d & 1) ? __cosf(ang) : __sinf(ang);
        float v = emb[(long long)t * Dd + d] + pe;
        x[(long long)row * Dd + d]  = v;
        xb[(long long)row * Dd + d] = (__bf16)v;
    }
}

// ---------------------------------------------------------------- WMMA GEMM
// C[M,N] = A[M,K](bf16) * B[K,N](bf16, pre-swizzled fragment-major) + bias.
// 256 threads = 8 waves (2x4), block tile 64(M) x 128(N), wave tile 32x32.
// A staged in LDS (reused by 4 n-wave columns); B fragments loaded straight
// from global as aligned 32B-per-lane vectors (L2-resident weights).
__global__ __launch_bounds__(256)
void k_gemm_bf16(const __bf16* __restrict__ A, const __bf16* __restrict__ Bsw,
                 const float* __restrict__ bias, const float* __restrict__ resid,
                 float* __restrict__ outF, __bf16* __restrict__ outB,
                 __bf16* __restrict__ outT, int M, int N, int K, int relu) {
    __shared__ alignas(16) __bf16 As[64 * 32];

    int tid = threadIdx.x, lane = tid & 31, w = tid >> 5;
    int wm = w & 1, wn = w >> 1;                 // 2 x 4 wave grid
    int mBlk = blockIdx.x * 64, nBlk = blockIdx.y * 128;
    int KS = K >> 5;
    int nt0 = (nBlk >> 4) + wn * 2;              // this wave's first n-tile

    const bf16x16* Bf = (const bf16x16*)Bsw;     // 32 frags (1KB) per block
    f32x8 acc[2][2] = {};

    for (int ks = 0; ks < KS; ++ks) {
        int k0 = ks * 32;
        // stage A tile (64x32): 8 bf16 (16B) per thread
        {
            int idx = tid * 8, r = idx >> 5, cA = idx & 31;
            int gr = mBlk + r; if (gr >= M) gr = M - 1;
            *(uint4*)(As + idx) = *(const uint4*)(A + (size_t)gr * K + k0 + cA);
            if (k0 + 32 < K)
                __builtin_prefetch((const void*)(A + (size_t)gr * K + k0 + 32 + cA), 0, 1);
        }
        __syncthreads();

        bf16x16 a0 = load_fragA(As, 32, wm * 32,      lane);
        bf16x16 a1 = load_fragA(As, 32, wm * 32 + 16, lane);
        bf16x16 b0 = Bf[((size_t)nt0 * KS + ks) * 32 + lane];
        bf16x16 b1 = Bf[((size_t)(nt0 + 1) * KS + ks) * 32 + lane];
        if (ks + 1 < KS) {
            __builtin_prefetch((const void*)&Bf[((size_t)nt0 * KS + ks + 1) * 32 + lane], 0, 1);
            __builtin_prefetch((const void*)&Bf[((size_t)(nt0 + 1) * KS + ks + 1) * 32 + lane], 0, 1);
        }

        acc[0][0] = __builtin_amdgcn_wmma_f32_16x16x32_bf16(false, a0, false, b0, (short)0, acc[0][0], false, false);
        acc[0][1] = __builtin_amdgcn_wmma_f32_16x16x32_bf16(false, a0, false, b1, (short)0, acc[0][1], false, false);
        acc[1][0] = __builtin_amdgcn_wmma_f32_16x16x32_bf16(false, a1, false, b0, (short)0, acc[1][0], false, false);
        acc[1][1] = __builtin_amdgcn_wmma_f32_16x16x32_bf16(false, a1, false, b1, (short)0, acc[1][1], false, false);
        __syncthreads();
    }

    // epilogue (C layout: lane -> n = lane&15, VGPR r -> m = r + 8*(lane>>4))
    int half = lane >> 4, col = lane & 15;
#pragma unroll
    for (int i = 0; i < 2; ++i)
#pragma unroll
        for (int j = 0; j < 2; ++j) {
            int n = nBlk + wn * 32 + j * 16 + col;
            float bv = bias ? bias[n] : 0.f;
#pragma unroll
            for (int r = 0; r < 8; ++r) {
                int m = mBlk + wm * 32 + i * 16 + r + half * 8;
                float v = acc[i][j][r] + bv;
                if (relu) v = fmaxf(v, 0.f);
                if (outT)                    // transposed store, padded: no guard
                    outT[(size_t)n * MPAD + m] = (__bf16)v;
                if (m < M) {
                    if (resid) v += resid[(size_t)m * N + n];
                    if (outF)  outF[(size_t)m * N + n] = v;
                    if (outB)  outB[(size_t)m * N + n] = (__bf16)v;
                }
            }
        }
}

// ---------------------------------------------------------------- fused attention
// One workgroup per (128 query rows, head, batch). 8 waves; wave = 16 q rows.
// Flash-style online softmax; QK^T and P*V via WMMA bf16.
// K arrives key-major [B*S][D]; V arrives transposed [D-slot][MPAD] so that
// both K and V B-fragments are contiguous 16B chunks in LDS.
__global__ __launch_bounds__(256)
void k_attn(const __bf16* __restrict__ Q, const __bf16* __restrict__ Km,
            const __bf16* __restrict__ VT, __bf16* __restrict__ O) {
    __shared__ alignas(16) __bf16 Ks[64 * 64];   // [key][dim]
    __shared__ alignas(16) __bf16 Vt[64 * 64];   // [dim][key]
    __shared__ alignas(16) __bf16 Ps[8][16 * 64];

    int tid = threadIdx.x, lane = tid & 31, w = tid >> 5;
    int b = blockIdx.z, h = blockIdx.y;
    int qbase = blockIdx.x * 128 + w * 16;
    int half = lane >> 4, col = lane & 15;
    const float scale = 0.125f;   // 1/sqrt(64)

    // Q fragments (K-dim 64 => two 32-wide slabs), rows clamped
    int qm = qbase + col; if (qm >= Ss) qm = Ss - 1;
    const __bf16* qrow = Q + ((size_t)(b * Ss + qm)) * Dd + h * HD;
    bf16x16 qa[2];
#pragma unroll
    for (int f = 0; f < 2; ++f)
#pragma unroll
        for (int e = 0; e < 16; ++e) qa[f][e] = qrow[f * 32 + frag_k(e, half)];

    f32x8 octx[4] = {};
    float Mrow[8], Lrow[8];
#pragma unroll
    for (int r = 0; r < 8; ++r) { Mrow[r] = -1e30f; Lrow[r] = 0.f; }

    for (int j = 0; j < Ss; j += 64) {
        // stage K (key-major, per-key clamp) and V (dim-major, padded cols)
        {
            int r = tid >> 2, c = (tid & 3) * 16;
            int key = j + r; if (key >= Ss) key = Ss - 1;
            const uint4* ks = (const uint4*)(Km + ((size_t)(b * Ss + key)) * Dd + h * HD + c);
            uint4* kd = (uint4*)(Ks + r * 64 + c);
            kd[0] = ks[0]; kd[1] = ks[1];
            // V: row = head-dim slot, cols = keys j+c .. j+c+15 (padding finite)
            const uint4* vs = (const uint4*)(VT + ((size_t)(h * HD + r)) * MPAD + b * Ss + j + c);
            uint4* vd = (uint4*)(Vt + r * 64 + c);
            vd[0] = vs[0]; vd[1] = vs[1];
        }
        __syncthreads();

        // scores: 4 n-tiles of 16 keys, K-dim 64 (2 WMMA each)
        f32x8 st[4];
#pragma unroll
        for (int t = 0; t < 4; ++t) {
            f32x8 s = {};
#pragma unroll
            for (int f = 0; f < 2; ++f) {
                bf16x16 bk;   // B(k,n) = K[t*16+n][k] : contiguous per lane
#pragma unroll
                for (int e = 0; e < 16; ++e)
                    bk[e] = Ks[(t * 16 + col) * 64 + f * 32 + frag_k(e, half)];
                s = __builtin_amdgcn_wmma_f32_16x16x32_bf16(false, qa[f], false, bk, (short)0, s, false, false);
            }
            int key = j + t * 16 + col;
#pragma unroll
            for (int r = 0; r < 8; ++r)
                s[r] = (key < Ss) ? s[r] * scale : -1e30f;
            st[t] = s;
        }

        // online softmax: row reductions across 16 lanes (wave32, width 16)
        float nm[8], alpha[8];
#pragma unroll
        for (int r = 0; r < 8; ++r) {
            float mx = fmaxf(fmaxf(st[0][r], st[1][r]), fmaxf(st[2][r], st[3][r]));
#pragma unroll
            for (int off = 8; off >= 1; off >>= 1) mx = fmaxf(mx, __shfl_xor(mx, off, 16));
            nm[r] = fmaxf(Mrow[r], mx);
            alpha[r] = __expf(Mrow[r] - nm[r]);
        }
#pragma unroll
        for (int r = 0; r < 8; ++r) {
            float sum = 0.f;
#pragma unroll
            for (int t = 0; t < 4; ++t) {
                float p = __expf(st[t][r] - nm[r]);
                st[t][r] = p;
                sum += p;
            }
#pragma unroll
            for (int off = 8; off >= 1; off >>= 1) sum += __shfl_xor(sum, off, 16);
            Lrow[r] = Lrow[r] * alpha[r] + sum;
            Mrow[r] = nm[r];
        }

        // P -> LDS (re-layout C-frag -> A-frag), wave-private region
#pragma unroll
        for (int t = 0; t < 4; ++t)
#pragma unroll
            for (int r = 0; r < 8; ++r)
                Ps[w][(r + half * 8) * 64 + t * 16 + col] = (__bf16)st[t][r];
        __syncthreads();

        // rescale running context, then ctx += P(16x64) * V(64x64)
#pragma unroll
        for (int nt = 0; nt < 4; ++nt)
#pragma unroll
            for (int r = 0; r < 8; ++r) octx[nt][r] *= alpha[r];

        bf16x16 pa[2];
#pragma unroll
        for (int f = 0; f < 2; ++f)
#pragma unroll
            for (int e = 0; e < 16; ++e)
                pa[f][e] = Ps[w][col * 64 + f * 32 + frag_k(e, half)];

#pragma unroll
        for (int nt = 0; nt < 4; ++nt)
#pragma unroll
            for (int f = 0; f < 2; ++f) {
                bf16x16 vb;   // B(k,n) = Vt[n][k] : contiguous per lane
#pragma unroll
                for (int e = 0; e < 16; ++e)
                    vb[e] = Vt[(nt * 16 + col) * 64 + f * 32 + frag_k(e, half)];
                octx[nt] = __builtin_amdgcn_wmma_f32_16x16x32_bf16(false, pa[f], false, vb, (short)0, octx[nt], false, false);
            }
        __syncthreads();   // protect Ks/Vt/Ps before next iteration's staging
    }

    // normalize and store context into [B,S,D] at head slot
#pragma unroll
    for (int nt = 0; nt < 4; ++nt)
#pragma unroll
        for (int r = 0; r < 8; ++r) {
            int m = qbase + r + half * 8;
            if (m < Ss) {
                float l = Lrow[r];
                float v = octx[nt][r] * (l > 0.f ? 1.f / l : 0.f);
                O[((size_t)(b * Ss + m)) * Dd + h * HD + nt * 16 + col] = (__bf16)v;
            }
        }
}

// ---------------------------------------------------------------- layernorm
// one wave per row of 512; 16 elems/lane; wave32 shfl reductions
__global__ __launch_bounds__(256)
void k_layernorm(const float* __restrict__ in, const float* __restrict__ g,
                 const float* __restrict__ beta, float* __restrict__ outF,
                 __bf16* __restrict__ outB, int rows) {
    int lane = threadIdx.x & 31, w = threadIdx.x >> 5;
    int row = blockIdx.x * 8 + w;
    if (row >= rows) return;
    const float* p = in + (size_t)row * Dd;
    float vals[16];
    float sum = 0.f;
#pragma unroll
    for (int i = 0; i < 16; ++i) { vals[i] = p[lane + i * 32]; sum += vals[i]; }
#pragma unroll
    for (int off = 16; off >= 1; off >>= 1) sum += __shfl_xor(sum, off, 32);
    float mu = sum * (1.f / Dd);
    float var = 0.f;
#pragma unroll
    for (int i = 0; i < 16; ++i) { float d = vals[i] - mu; var += d * d; }
#pragma unroll
    for (int off = 16; off >= 1; off >>= 1) var += __shfl_xor(var, off, 32);
    float inv = rsqrtf(var * (1.f / Dd) + 1e-5f);
#pragma unroll
    for (int i = 0; i < 16; ++i) {
        int d = lane + i * 32;
        float o = (vals[i] - mu) * inv * g[d] + beta[d];
        outF[(size_t)row * Dd + d] = o;
        outB[(size_t)row * Dd + d] = (__bf16)o;
    }
}

// ---------------------------------------------------------------- row softmax (in place)
__global__ __launch_bounds__(256)
void k_softmax(float* __restrict__ x, int cols) {
    __shared__ float red[8];
    float* p = x + (size_t)blockIdx.x * cols;
    int lane = threadIdx.x & 31, w = threadIdx.x >> 5;

    float mx = -1e30f;
    for (int c = threadIdx.x; c < cols; c += 256) mx = fmaxf(mx, p[c]);
#pragma unroll
    for (int off = 16; off >= 1; off >>= 1) mx = fmaxf(mx, __shfl_xor(mx, off, 32));
    if (lane == 0) red[w] = mx;
    __syncthreads();
    mx = red[0];
#pragma unroll
    for (int i = 1; i < 8; ++i) mx = fmaxf(mx, red[i]);
    __syncthreads();

    float sum = 0.f;
    for (int c = threadIdx.x; c < cols; c += 256) {
        float e = __expf(p[c] - mx);
        p[c] = e;
        sum += e;
    }
#pragma unroll
    for (int off = 16; off >= 1; off >>= 1) sum += __shfl_xor(sum, off, 32);
    if (lane == 0) red[w] = sum;
    __syncthreads();
    float tot = 0.f;
#pragma unroll
    for (int i = 0; i < 8; ++i) tot += red[i];
    float inv = 1.f / tot;
    for (int c = threadIdx.x; c < cols; c += 256) p[c] *= inv;
}

// ---------------------------------------------------------------- host driver
extern "C" void kernel_launch(void* const* d_in, const int* in_sizes, int n_in,
                              void* d_out, int out_size, void* d_ws, size_t ws_size,
                              hipStream_t stream) {
    (void)in_sizes; (void)n_in; (void)out_size; (void)ws_size;
    const int*   tokens = (const int*)d_in[0];
    const float* emb  = (const float*)d_in[1];
    const float* wq   = (const float*)d_in[2];
    const float* bq   = (const float*)d_in[3];
    const float* wk   = (const float*)d_in[4];
    const float* bk   = (const float*)d_in[5];
    const float* wv   = (const float*)d_in[6];
    const float* bv   = (const float*)d_in[7];
    const float* wo   = (const float*)d_in[8];
    const float* bo   = (const float*)d_in[9];
    const float* ln1g = (const float*)d_in[10];
    const float* ln1b = (const float*)d_in[11];
    const float* w1   = (const float*)d_in[12];
    const float* b1   = (const float*)d_in[13];
    const float* w2   = (const float*)d_in[14];
    const float* b2   = (const float*)d_in[15];
    const float* ln2g = (const float*)d_in[16];
    const float* ln2b = (const float*)d_in[17];
    const float* fcw  = (const float*)d_in[18];
    const float* fcb  = (const float*)d_in[19];
    float* out = (float*)d_out;

    char* ws = (char*)d_ws;
    size_t off = 0;
    auto bump = [&](size_t bytes) -> char* {
        char* p = ws + off;
        off += (bytes + 255) & ~(size_t)255;
        return p;
    };
    float*  x    = (float*)bump((size_t)ROWS * Dd * 4);
    float*  t0   = (float*)bump((size_t)ROWS * Dd * 4);
    __bf16* xb   = (__bf16*)bump((size_t)ROWS * Dd * 2);
    __bf16* qb   = (__bf16*)bump((size_t)ROWS * Dd * 2);
    __bf16* kb2  = (__bf16*)bump((size_t)ROWS * Dd * 2);
    __bf16* vt   = (__bf16*)bump((size_t)Dd * MPAD * 2);     // transposed V
    __bf16* attb = (__bf16*)bump((size_t)ROWS * Dd * 2);
    __bf16* hb   = (__bf16*)bump((size_t)ROWS * FFd * 2);
    __bf16* wqb  = (__bf16*)bump((size_t)Ll * Dd * Dd * 2);
    __bf16* wkb  = (__bf16*)bump((size_t)Ll * Dd * Dd * 2);
    __bf16* wvb  = (__bf16*)bump((size_t)Ll * Dd * Dd * 2);
    __bf16* wob  = (__bf16*)bump((size_t)Ll * Dd * Dd * 2);
    __bf16* w1b  = (__bf16*)bump((size_t)Ll * Dd * FFd * 2);
    __bf16* w2b  = (__bf16*)bump((size_t)Ll * FFd * Dd * 2);
    __bf16* fcwb = (__bf16*)bump((size_t)Dd * Vv * 2);

    // pack every weight matrix into WMMA-B fragment-major bf16 layout
    auto pack = [&](const float* s, __bf16* d, int K, int N) {
        long long total = (long long)K * N;
        long long blocks = (total + 255) / 256;
        if (blocks > 8192) blocks = 8192;
        hipLaunchKernelGGL(k_pack_b, dim3((unsigned)blocks), dim3(256), 0, stream, s, d, K, N);
    };
    for (int i = 0; i < Ll; ++i) {
        pack(wq + (size_t)i * Dd * Dd,  wqb + (size_t)i * Dd * Dd,  Dd, Dd);
        pack(wk + (size_t)i * Dd * Dd,  wkb + (size_t)i * Dd * Dd,  Dd, Dd);
        pack(wv + (size_t)i * Dd * Dd,  wvb + (size_t)i * Dd * Dd,  Dd, Dd);
        pack(wo + (size_t)i * Dd * Dd,  wob + (size_t)i * Dd * Dd,  Dd, Dd);
        pack(w1 + (size_t)i * Dd * FFd, w1b + (size_t)i * Dd * FFd, Dd, FFd);
        pack(w2 + (size_t)i * FFd * Dd, w2b + (size_t)i * FFd * Dd, FFd, Dd);
    }
    pack(fcw, fcwb, Dd, Vv);

    hipLaunchKernelGGL(k_embed, dim3(ROWS), dim3(128), 0, stream, tokens, emb, x, xb);

    auto gemm = [&](const __bf16* A, const __bf16* Bw, const float* bias,
                    const float* resid, float* oF, __bf16* oB, __bf16* oT,
                    int M, int N, int K, int relu) {
        dim3 grid((M + 63) / 64, N / 128);
        hipLaunchKernelGGL(k_gemm_bf16, grid, dim3(256), 0, stream,
                           A, Bw, bias, resid, oF, oB, oT, M, N, K, relu);
    };

    for (int i = 0; i < Ll; ++i) {
        const __bf16* wqi = wqb + (size_t)i * Dd * Dd;
        const __bf16* wki = wkb + (size_t)i * Dd * Dd;
        const __bf16* wvi = wvb + (size_t)i * Dd * Dd;
        const __bf16* woi = wob + (size_t)i * Dd * Dd;
        const __bf16* w1i = w1b + (size_t)i * Dd * FFd;
        const __bf16* w2i = w2b + (size_t)i * FFd * Dd;

        gemm(xb, wqi, bq + i * Dd, nullptr, nullptr, qb,  nullptr, ROWS, Dd, Dd, 0);
        gemm(xb, wki, bk + i * Dd, nullptr, nullptr, kb2, nullptr, ROWS, Dd, Dd, 0);
        gemm(xb, wvi, bv + i * Dd, nullptr, nullptr, nullptr, vt,  ROWS, Dd, Dd, 0);

        hipLaunchKernelGGL(k_attn, dim3((Ss + 127) / 128, Hh, Bb), dim3(256), 0, stream,
                           qb, kb2, vt, attb);

        gemm(attb, woi, bo + i * Dd, x, t0, nullptr, nullptr, ROWS, Dd, Dd, 0);
        hipLaunchKernelGGL(k_layernorm, dim3((ROWS + 7) / 8), dim3(256), 0, stream,
                           t0, ln1g + i * Dd, ln1b + i * Dd, x, xb, ROWS);

        gemm(xb, w1i, b1 + i * FFd, nullptr, nullptr, hb, nullptr, ROWS, FFd, Dd, 1);
        gemm(hb, w2i, b2 + i * Dd, x, t0, nullptr, nullptr, ROWS, Dd, FFd, 0);
        hipLaunchKernelGGL(k_layernorm, dim3((ROWS + 7) / 8), dim3(256), 0, stream,
                           t0, ln2g + i * Dd, ln2b + i * Dd, x, xb, ROWS);
    }

    gemm(xb, fcwb, fcb, nullptr, out, nullptr, nullptr, ROWS, Vv, Dd, 0);
    hipLaunchKernelGGL(k_softmax, dim3(ROWS), dim3(256), 0, stream, out, Vv);
}
